// TorchKCnn_65137474011868
// MI455X (gfx1250) — matrix operand order repaired and checked by reading, not compiled, not run on hardware
//
#include <hip/hip_runtime.h>

typedef __attribute__((ext_vector_type(16))) __bf16 v16bf;
typedef __attribute__((ext_vector_type(8)))  float  v8f;

#define B_  256
#define S_  512
#define K_  32
#define F_  64
#define H1_ 512
#define H2_ 256
#define MT  64   // M rows per block

union Frag { uint4 q[2]; v16bf v; };

__device__ __forceinline__ unsigned short f2bf(float f) {
  unsigned u = __float_as_uint(f);
  u += 0x7FFFu + ((u >> 16) & 1u);        // round-to-nearest-even
  return (unsigned short)(u >> 16);
}
__device__ __forceinline__ unsigned pack2(float a, float b) {
  return (unsigned)f2bf(a) | ((unsigned)f2bf(b) << 16);
}

// CDNA5 async global->LDS 16B copy (ASYNCcnt). LDS address = low 32 bits of
// the generic pointer (aperture rule, ISA 10.2).
__device__ __forceinline__ void async_cp16(const void* g, void* l) {
  unsigned lds = (unsigned)(unsigned long long)l;
  asm volatile("global_load_async_to_lds_b128 %0, %1, off"
               :: "v"(lds), "v"(g) : "memory");
}
__device__ __forceinline__ void wait_async() {
  asm volatile("s_wait_asynccnt 0x0" ::: "memory");
}

// ---------------- workspace layout (bytes) ----------------
#define WS_M     0                         // B_*H2_ f32 (max buffer)   256 KB
#define WS_XB    (262144)                  // x as bf16                 16 MB
#define WS_DB    (WS_XB + 16777216)        // distances bf16            32 KB
#define WS_W1    (WS_DB + 32768)           // W1 bf16 [65][512][32]     2.03 MB
#define WS_W2    (WS_W1 + 2129920)         // W2 bf16 [16][256][32]     256 KB

// ---------------- prep kernels (one-time per launch) ----------------
__global__ void __launch_bounds__(256) prep_x(const float* __restrict__ x,
                                              unsigned short* __restrict__ xb) {
  const int i = blockIdx.x * 256 + threadIdx.x;   // 8 floats per thread
  const float4 f0 = reinterpret_cast<const float4*>(x)[i * 2];
  const float4 f1 = reinterpret_cast<const float4*>(x)[i * 2 + 1];
  uint4 o;
  o.x = pack2(f0.x, f0.y); o.y = pack2(f0.z, f0.w);
  o.z = pack2(f1.x, f1.y); o.w = pack2(f1.z, f1.w);
  reinterpret_cast<uint4*>(xb)[i] = o;
}

__global__ void __launch_bounds__(256) prep_dist(const float* __restrict__ d,
                                                 unsigned short* __restrict__ db) {
  const int i = blockIdx.x * 256 + threadIdx.x;
  db[i] = f2bf(d[i]);
}

// wt1[kc][n][j] = bf16(W1[g(kc,j)][n]); g reorders 2080 rows into
// 64 chunks of neighbor-features (32 each) + 1 chunk of distance rows.
__global__ void __launch_bounds__(256) prep_w1(const float* __restrict__ W1,
                                               unsigned short* __restrict__ wt1) {
  const int e  = blockIdx.x * 256 + threadIdx.x;  // 65*512*32 elements
  const int kc = e >> 14;
  const int r  = e & 16383;
  const int j  = r >> 9;
  const int n  = r & 511;
  const int g  = (kc < 64) ? ((kc >> 1) * 65 + (kc & 1) * 32 + j) : (j * 65 + 64);
  wt1[kc * 16384 + n * 32 + j] = f2bf(W1[(size_t)g * H1_ + n]);
}

// wt2[kki][n][j] = bf16(W2[kki*32+j][n]),  kki = 0..15
__global__ void __launch_bounds__(256) prep_w2(const float* __restrict__ W2,
                                               unsigned short* __restrict__ wt2) {
  const int e   = blockIdx.x * 256 + threadIdx.x; // 16*256*32 elements
  const int kki = e >> 13;
  const int r   = e & 8191;
  const int j   = r >> 8;
  const int n   = r & 255;
  wt2[kki * 8192 + n * 32 + j] = f2bf(W2[(size_t)(kki * 32 + j) * H2_ + n]);
}

__global__ void zero_m(int* __restrict__ m) {
  m[blockIdx.x * 256 + threadIdx.x] = 0;
}

// ---------------- fused main kernel ----------------
// gather -> GEMM1(2080x512,bf16 WMMA) -> relu -> GEMM2(512x256) -> col max
__global__ void __launch_bounds__(256)
kcnn_fused(const unsigned short* __restrict__ xb,
           const int* __restrict__ nbr,
           const unsigned short* __restrict__ distb,
           const unsigned short* __restrict__ wt1,
           const float* __restrict__ b1,
           const unsigned short* __restrict__ wt2,
           const float* __restrict__ b2,
           int* __restrict__ mOut)
{
  __shared__ int s_nbr[MT * K_];                               // 8 KB
  __shared__ alignas(16) unsigned short s_ab[2][MT * 32];      // 2x4 KB A tiles
  __shared__ alignas(16) unsigned short s_wb[2][64 * 32];      // 2x4 KB W1 tiles
  __shared__ alignas(16) unsigned short s_h1[MT * 64];         // 8 KB relu(h1)
  __shared__ alignas(16) unsigned short s_w2[H2_ * 32];        // 16 KB W2 tile
  __shared__ int s_colmax[H2_];                                // 1 KB

  const int tid  = threadIdx.x;
  const int wv   = tid >> 5;          // 8 waves (wave32)
  const int lane = tid & 31;
  const int hl   = (lane < 16) ? 0 : 1;
  const int l15  = lane & 15;
  const int b    = blockIdx.x >> 3;
  const int st   = (blockIdx.x & 7) * MT;

  for (int i = tid; i < MT * K_; i += 256)
    s_nbr[i] = nbr[(st + (i >> 5)) * K_ + (i & 31)];
  __syncthreads();

  // per-thread staging coordinates (16B granules)
  const int am = tid >> 2;            // A: row 0..63
  const int aj = (tid & 3) * 8;       // A: k offset 0/8/16/24

  v8f zero8 = {};
  v8f acc2[8];
#pragma unroll
  for (int i = 0; i < 8; ++i) acc2[i] = zero8;

  for (int n1c = 0; n1c < 8; ++n1c) {
    const int nbase = n1c * 64;
    v8f acc1[2]; acc1[0] = zero8; acc1[1] = zero8;

    // ---- issue chunk 0 stage (async), then pipeline 65 K-chunks ----
    {
      const int nb = s_nbr[am * K_ + 0];
      async_cp16(xb + ((size_t)b * S_ + nb) * F_ + aj, &s_ab[0][am * 32 + aj]);
      async_cp16(wt1 + nbase * 32 + tid * 8, &s_wb[0][tid * 8]);
    }
    wait_async();
    __syncthreads();

    for (int kc = 0; kc < 65; ++kc) {
      const int cur = kc & 1;
      // stage chunk kc+1 into the other buffer (overlaps with WMMA below)
      if (kc < 64) {
        const int nxt = kc + 1;
        const unsigned short* asrc;
        if (nxt < 64) {
          const int nb = s_nbr[am * K_ + (nxt >> 1)];
          asrc = xb + ((size_t)b * S_ + nb) * F_ + (nxt & 1) * 32 + aj;
        } else {
          asrc = distb + (st + am) * K_ + aj;
        }
        async_cp16(asrc, &s_ab[cur ^ 1][am * 32 + aj]);
        async_cp16(wt1 + (size_t)nxt * 16384 + nbase * 32 + tid * 8,
                   &s_wb[cur ^ 1][tid * 8]);
      }
      // ---- GEMM1 step on current buffers ----
      {
        const int nt = wv & 3;
        Frag bf;
        const unsigned short* wp = &s_wb[cur][(nt * 16 + l15) * 32 + hl * 8];
        bf.q[0] = *reinterpret_cast<const uint4*>(wp);
        bf.q[1] = *reinterpret_cast<const uint4*>(wp + 16);
#pragma unroll
        for (int t = 0; t < 2; ++t) {
          const int mt = (wv >> 2) * 2 + t;
          Frag af;
          const unsigned short* ap = &s_ab[cur][(mt * 16 + l15) * 32 + hl * 8];
          af.q[0] = *reinterpret_cast<const uint4*>(ap);
          af.q[1] = *reinterpret_cast<const uint4*>(ap + 16);
          acc1[t] = __builtin_amdgcn_wmma_f32_16x16x32_bf16(
              false, af.v, false, bf.v, (short)0, acc1[t], false, false);
        }
      }
      wait_async();
      __syncthreads();
    }

    // ---- bias + relu -> bf16 h1 chunk in LDS ----
    {
      const int nt = wv & 3;
      const float bias = b1[nbase + nt * 16 + l15];
#pragma unroll
      for (int t = 0; t < 2; ++t) {
        const int mt = (wv >> 2) * 2 + t;
#pragma unroll
        for (int r = 0; r < 8; ++r) {
          float v = fmaxf(acc1[t][r] + bias, 0.0f);
          s_h1[(mt * 16 + r + hl * 8) * 64 + nt * 16 + l15] = f2bf(v);
        }
      }
    }
    __syncthreads();

    // ---- GEMM2 partial: C2 += h1c(64x64) @ W2[nbase:nbase+64, :] ----
    for (int kk = 0; kk < 2; ++kk) {
      { // async-stage 16 KB W2 tile for k-chunk kki = n1c*2+kk
        const unsigned short* src = wt2 + (size_t)(n1c * 2 + kk) * 8192 + tid * 8;
#pragma unroll
        for (int rep = 0; rep < 4; ++rep)
          async_cp16(src + rep * 2048, &s_w2[tid * 8 + rep * 2048]);
      }
      wait_async();
      __syncthreads();
      {
        Frag bfr[2];
#pragma unroll
        for (int q = 0; q < 2; ++q) {
          const unsigned short* wp =
              &s_w2[((wv * 2 + q) * 16 + l15) * 32 + hl * 8];
          bfr[q].q[0] = *reinterpret_cast<const uint4*>(wp);
          bfr[q].q[1] = *reinterpret_cast<const uint4*>(wp + 16);
        }
#pragma unroll
        for (int mt = 0; mt < 4; ++mt) {
          Frag af;
          const unsigned short* ap =
              &s_h1[(mt * 16 + l15) * 64 + kk * 32 + hl * 8];
          af.q[0] = *reinterpret_cast<const uint4*>(ap);
          af.q[1] = *reinterpret_cast<const uint4*>(ap + 16);
#pragma unroll
          for (int q = 0; q < 2; ++q) {
            acc2[mt * 2 + q] = __builtin_amdgcn_wmma_f32_16x16x32_bf16(
                false, af.v, false, bfr[q].v, (short)0, acc2[mt * 2 + q],
                false, false);
          }
        }
      }
      __syncthreads();
    }
  }

  // ---- b2 + block-local column max (relu folded via 0-init), then global ----
  s_colmax[tid] = 0;
  __syncthreads();
#pragma unroll
  for (int i = 0; i < 8; ++i) {
    const int mt = i >> 1;
    const int q  = i & 1;
    const int col = (wv * 2 + q) * 16 + l15;
    const float bias2 = b2[col];
    float mx = 0.0f;
#pragma unroll
    for (int r = 0; r < 8; ++r) mx = fmaxf(mx, acc2[mt * 2 + q][r] + bias2);
    atomicMax(&s_colmax[col], __float_as_int(mx));  // vals >= 0: int order ok
  }
  __syncthreads();
  atomicMax(mOut + b * H2_ + tid, s_colmax[tid]);
}

// out[b] = m[b,:] . Wd + bd
__global__ void __launch_bounds__(256)
head_kernel(const int* __restrict__ m, const float* __restrict__ Wd,
            const float* __restrict__ bd, float* __restrict__ out) {
  __shared__ float red[256];
  const int b = blockIdx.x, t = threadIdx.x;
  red[t] = __int_as_float(m[b * H2_ + t]) * Wd[t];
  __syncthreads();
  for (int s = 128; s > 0; s >>= 1) {
    if (t < s) red[t] += red[t + s];
    __syncthreads();
  }
  if (t == 0) out[b] = red[0] + bd[0];
}

extern "C" void kernel_launch(void* const* d_in, const int* in_sizes, int n_in,
                              void* d_out, int out_size, void* d_ws, size_t ws_size,
                              hipStream_t stream) {
  const float* x    = (const float*)d_in[0];
  const int*   nbr  = (const int*)d_in[1];
  const float* dist = (const float*)d_in[2];
  const float* W1   = (const float*)d_in[3];
  const float* b1   = (const float*)d_in[4];
  const float* W2   = (const float*)d_in[5];
  const float* b2   = (const float*)d_in[6];
  const float* Wd   = (const float*)d_in[7];
  const float* bd   = (const float*)d_in[8];
  (void)in_sizes; (void)n_in; (void)out_size; (void)ws_size;

  char* ws = (char*)d_ws;
  int*            m    = (int*)(ws + WS_M);
  unsigned short* xb   = (unsigned short*)(ws + WS_XB);
  unsigned short* db   = (unsigned short*)(ws + WS_DB);
  unsigned short* wt1  = (unsigned short*)(ws + WS_W1);
  unsigned short* wt2  = (unsigned short*)(ws + WS_W2);

  // one-time converts (bf16, fragment-layout) + max-buffer zero
  prep_x   <<<(B_ * S_ * F_) / (256 * 8), 256, 0, stream>>>(x, xb);
  prep_dist<<<(S_ * K_) / 256,            256, 0, stream>>>(dist, db);
  prep_w1  <<<(65 * H1_ * 32) / 256,      256, 0, stream>>>(W1, wt1);
  prep_w2  <<<(16 * H2_ * 32) / 256,      256, 0, stream>>>(W2, wt2);
  zero_m   <<<B_, H2_, 0, stream>>>(m);

  kcnn_fused<<<B_ * (S_ / MT), 256, 0, stream>>>(xb, nbr, db, wt1, b1, wt2, b2, m);
  head_kernel<<<B_, H2_, 0, stream>>>(m, Wd, bd, (float*)d_out);
}
